// MultiHeadAttention_23218593202571
// MI455X (gfx1250) — compile-verified, matrix operand order checked by reading, not run on hardware
//
#include <hip/hip_runtime.h>

// ---------------------------------------------------------------------------
// MI455X (gfx1250) MHA forward.
// Compute-bound (~69 GFLOP vs ~100MB HBM => ~4.3us memory floor at 23.3 TB/s),
// so all matmuls run on v_wmma_f32_16x16x32_f16 (fp32 accumulate), softmax in
// fp32 VALU. wave32; 256-thread blocks = 8 waves. GEMM tiles double-buffered
// through LDS with global_load_async_to_lds_b128 (ASYNCcnt) when available.
// ---------------------------------------------------------------------------

typedef __attribute__((ext_vector_type(16))) _Float16 v16h;
typedef __attribute__((ext_vector_type(8)))  _Float16 v8h;
typedef __attribute__((ext_vector_type(4)))  _Float16 v4h;
typedef __attribute__((ext_vector_type(8)))  float    v8f;
typedef __attribute__((ext_vector_type(4)))  int      v4i;

// pointer-to-(addrspace-qualified v4i): attribute binds to the pointee
typedef __attribute__((address_space(1))) v4i* gbl_v4i_ptr;
typedef __attribute__((address_space(3))) v4i* lds_v4i_ptr;

#define DIM_MODEL 1024
#define SEQ       2048
#define HEADS     16
#define DH        64

#if defined(__AMDGCN__) && __has_builtin(__builtin_amdgcn_global_load_async_to_lds_b128) && \
    __has_builtin(__builtin_amdgcn_s_wait_asynccnt)
#define USE_ASYNC 1
#else
#define USE_ASYNC 0
#endif

// 16-byte global->LDS copy. Async (no VGPR round trip, ASYNCcnt-tracked) when
// the gfx1250 builtin exists; falls back to a b128 load/store pair.
__device__ __forceinline__ void cp16_g2l(const _Float16* g, _Float16* l) {
#if USE_ASYNC
  __builtin_amdgcn_global_load_async_to_lds_b128(
      (gbl_v4i_ptr)(void*)g, (lds_v4i_ptr)(void*)l, 0, 0);
#else
  *(v8h*)l = *(const v8h*)g;
#endif
}

template <int N>
__device__ __forceinline__ void wait_async() {
#if USE_ASYNC
  __builtin_amdgcn_s_wait_asynccnt(N);
#endif
}

// Fragment loader for A/B operands of V_WMMA_F32_16X16X32_F16.
// Covers a 16(row) x 32(k) tile at p, leading dim ld (halfs).
// ISA 16-bit A layout: lane l (0..15): row=l, K={0..7,16..23};
//                      lane l+16:      row=l, K={8..15,24..31}.
// B operand (K x N) uses the identical pattern on row-major [N][K] storage.
__device__ __forceinline__ v16h load_frag(const _Float16* p, int ld) {
  const int lane = threadIdx.x & 31;
  const _Float16* q = p + (lane & 15) * ld + ((lane >> 4) << 3);
  v8h lo = *(const v8h*)(q);
  v8h hi = *(const v8h*)(q + 16);
  v16h r;
#pragma unroll
  for (int i = 0; i < 8; ++i) { r[i] = lo[i]; r[i + 8] = hi[i]; }
  return r;
}

__device__ __forceinline__ void store_out(float* p, float v)     { *p = v; }
__device__ __forceinline__ void store_out(_Float16* p, float v)  { *p = (_Float16)v; }

// ---------------------------------------------------------------------------
__global__ void cvt_f32_f16(const float* __restrict__ in, _Float16* __restrict__ out, int n4) {
  int i = blockIdx.x * blockDim.x + threadIdx.x;
  if (i < n4) {
    float4 v = ((const float4*)in)[i];
    v4h o;
    o[0] = (_Float16)v.x; o[1] = (_Float16)v.y;
    o[2] = (_Float16)v.z; o[3] = (_Float16)v.w;
    ((v4h*)out)[i] = o;
  }
}

// ---------------------------------------------------------------------------
// C[M,N] = A[M,K] @ Bw[N,K]^T + bias[N].  128x128 block tile, k-step 32,
// double-buffered LDS fed by async copies. 8 waves in 2(M) x 4(N) layout;
// each wave owns a 64x32 tile = 4x2 WMMA tiles.
template <typename OutT>
__global__ __launch_bounds__(256)
void gemm_bias(const _Float16* __restrict__ A, const _Float16* __restrict__ Bw,
               const float* __restrict__ bias, OutT* __restrict__ C,
               int M, int N, int K) {
  __shared__ __align__(16) _Float16 sA[2][128 * 40];  // 32 + 8 pad halfs
  __shared__ __align__(16) _Float16 sB[2][128 * 40];

  const int m0 = blockIdx.y * 128, n0 = blockIdx.x * 128;
  const int wave = threadIdx.x >> 5;
  const int wm = wave >> 2, wn = wave & 3;

  // Per-thread tile-fill coordinates: 512 x 16B chunks per matrix, 2 each.
  const int c0 = threadIdx.x,        r0 = c0 >> 2, cc0 = (c0 & 3) * 8;
  const int c1 = threadIdx.x + 256,  r1 = c1 >> 2, cc1 = (c1 & 3) * 8;

  v8f acc[4][2];
#pragma unroll
  for (int mi = 0; mi < 4; ++mi)
#pragma unroll
    for (int ni = 0; ni < 2; ++ni) acc[mi][ni] = (v8f)0.0f;

  // Each thread issues exactly 4 copies per tile (2 for A, 2 for B); async
  // completion is in-order, so s_wait_asynccnt<=4 == "current tile landed".
#define ISSUE_TILE(buf, kt)                                                         \
  do {                                                                              \
    cp16_g2l(A  + (size_t)(m0 + r0) * K + (kt) + cc0, sA[buf] + r0 * 40 + cc0);     \
    cp16_g2l(Bw + (size_t)(n0 + r0) * K + (kt) + cc0, sB[buf] + r0 * 40 + cc0);     \
    cp16_g2l(A  + (size_t)(m0 + r1) * K + (kt) + cc1, sA[buf] + r1 * 40 + cc1);     \
    cp16_g2l(Bw + (size_t)(n0 + r1) * K + (kt) + cc1, sB[buf] + r1 * 40 + cc1);     \
  } while (0)

  ISSUE_TILE(0, 0);
  int buf = 0;
  for (int kt = 0; kt < K; kt += 32) {
    if (kt + 32 < K) { ISSUE_TILE(buf ^ 1, kt + 32); wait_async<4>(); }
    else             { wait_async<0>(); }
    __syncthreads();                         // current tile visible to all waves

    v16h af[4], bf[2];
#pragma unroll
    for (int mi = 0; mi < 4; ++mi) af[mi] = load_frag(sA[buf] + (wm * 64 + mi * 16) * 40, 40);
#pragma unroll
    for (int ni = 0; ni < 2; ++ni) bf[ni] = load_frag(sB[buf] + (wn * 32 + ni * 16) * 40, 40);
#pragma unroll
    for (int mi = 0; mi < 4; ++mi)
#pragma unroll
      for (int ni = 0; ni < 2; ++ni)
        acc[mi][ni] = __builtin_amdgcn_wmma_f32_16x16x32_f16(
            false, af[mi], false, bf[ni], (short)0, acc[mi][ni], false, false);

    __syncthreads();                         // reads done before buf is refilled
    buf ^= 1;
  }
#undef ISSUE_TILE

  const int lane = threadIdx.x & 31;
  const int cis = lane & 15, roff = (lane >> 4) * 8;  // C layout: vgpr r -> row r/r+8
#pragma unroll
  for (int mi = 0; mi < 4; ++mi)
#pragma unroll
    for (int ni = 0; ni < 2; ++ni) {
      int col = n0 + wn * 32 + ni * 16 + cis;
      float bv = bias[col];
#pragma unroll
      for (int r = 0; r < 8; ++r) {
        int row = m0 + wm * 64 + mi * 16 + roff + r;
        store_out(C + (size_t)row * N + col, acc[mi][ni][r] + bv);
      }
    }
}

// ---------------------------------------------------------------------------
// Flash attention, causal. Block = (b, h, 128 query rows); 8 waves, each wave
// owns 16 query rows. K/V processed in 64-row tiles. f16 WMMA, fp32 softmax.
// The 1/sqrt(Dh) scale is folded into Q at LDS-fill time; the causal
// compare/select runs only for tiles touching the wave's diagonal.
// qkv: f16 [B*S][3072] (Q|K|V by column block). attn: f16 [B*S][1024].
__global__ __launch_bounds__(256)
void attn_flash(const _Float16* __restrict__ qkv, _Float16* __restrict__ attn) {
  __shared__ __align__(16) _Float16 sQ [128 * 72];  // 128 x 64 (+8 pad)
  __shared__ __align__(16) _Float16 sK [ 64 * 72];  // 64 kpos x 64 dh
  __shared__ __align__(16) _Float16 sVt[ 64 * 72];  // 64 dh x 64 kpos (transposed)
  __shared__ __align__(16) _Float16 sP [128 * 72];  // per-wave 16x64 P slabs

  const int qt = blockIdx.x;            // 0..15  (128-row query tile)
  const int h  = blockIdx.y;            // 0..15
  const int b  = blockIdx.z;            // 0..1
  const int wave = threadIdx.x >> 5;
  const int lane = threadIdx.x & 31;
  const int cis = lane & 15, roff = (lane >> 4) * 8;

  const size_t brow0 = (size_t)b * SEQ;
  const int qoff = h * DH;
  const int koff = DIM_MODEL + h * DH;
  const int voff = 2 * DIM_MODEL + h * DH;

  // ---- load Q tile (128 x 64), scaled by 1/sqrt(64) ----
#pragma unroll
  for (int i = 0; i < 4; ++i) {
    int c = threadIdx.x + i * 256;              // 1024 x 16B chunks
    int r = c >> 3, cc = (c & 7) * 8;
    v8h q = *(const v8h*)(qkv + (brow0 + qt * 128 + r) * (3 * DIM_MODEL) + qoff + cc);
    q = q * (_Float16)0.125f;
    *(v8h*)(sQ + r * 72 + cc) = q;
  }
  __syncthreads();
  v16h qf[2];
  qf[0] = load_frag(sQ + wave * 16 * 72,      72);
  qf[1] = load_frag(sQ + wave * 16 * 72 + 32, 72);

  float m_run[8], l_run[8];
  v8f acc_o[4];
#pragma unroll
  for (int r = 0; r < 8; ++r) { m_run[r] = -3.0e38f; l_run[r] = 0.0f; }
#pragma unroll
  for (int no = 0; no < 4; ++no) acc_o[no] = (v8f)0.0f;

  const int row_base = qt * 128 + wave * 16 + roff;   // +r = query row in sequence
  const int nkt = 2 * (qt + 1);                       // causal: 64-row k tiles

  for (int kt = 0; kt < nkt; ++kt) {
    __syncthreads();                     // prior iteration's reads of sK/sVt done
    // ---- load K tile (async) + V tile (transposed into sVt) ----
#pragma unroll
    for (int i = 0; i < 2; ++i) {
      int c = threadIdx.x + i * 256;            // 512 x 16B chunks
      int r = c >> 3, cc = (c & 7) * 8;
      size_t grow = brow0 + kt * 64 + r;
      cp16_g2l(qkv + grow * (3 * DIM_MODEL) + koff + cc, sK + r * 72 + cc);
      v8h vv = *(const v8h*)(qkv + grow * (3 * DIM_MODEL) + voff + cc);
#pragma unroll
      for (int j = 0; j < 8; ++j) sVt[(cc + j) * 72 + r] = vv[j];
    }
    wait_async<0>();
    __syncthreads();

    // ---- S = Q K^T (16 x 64 per wave) ----
    v8f sacc[4];
#pragma unroll
    for (int ni = 0; ni < 4; ++ni) sacc[ni] = (v8f)0.0f;
#pragma unroll
    for (int ks = 0; ks < 2; ++ks)
#pragma unroll
      for (int ni = 0; ni < 4; ++ni) {
        v16h bf = load_frag(sK + ni * 16 * 72 + ks * 32, 72);
        sacc[ni] = __builtin_amdgcn_wmma_f32_16x16x32_f16(
            false, qf[ks], false, bf, (short)0, sacc[ni], false, false);
      }

    // ---- online softmax (fp32); mask only near the diagonal ----
    const bool tile_masked = (kt * 64 + 63) > (qt * 128 + wave * 16);
#pragma unroll
    for (int r = 0; r < 8; ++r) {
      float mold = m_run[r];
      float mn = mold;
#pragma unroll
      for (int ni = 0; ni < 4; ++ni) {
        float s = sacc[ni][r];
        if (tile_masked) {
          int col = kt * 64 + ni * 16 + cis;
          if (col > row_base + r) s = -3.0e38f;      // causal mask
          sacc[ni][r] = s;
        }
        mn = fmaxf(mn, s);
      }
#pragma unroll
      for (int off = 1; off < 16; off <<= 1) mn = fmaxf(mn, __shfl_xor(mn, off, 32));
      float corr = __expf(mold - mn);
      float rs = 0.0f;
#pragma unroll
      for (int ni = 0; ni < 4; ++ni) {
        float p = __expf(sacc[ni][r] - mn);
        sacc[ni][r] = p;
        rs += p;
      }
#pragma unroll
      for (int off = 1; off < 16; off <<= 1) rs += __shfl_xor(rs, off, 32);
      l_run[r] = l_run[r] * corr + rs;
      m_run[r] = mn;
#pragma unroll
      for (int no = 0; no < 4; ++no) acc_o[no][r] *= corr;
    }

    // ---- P (C-layout regs) -> per-wave LDS slab -> A-layout fragments ----
    _Float16* sPw = sP + wave * 16 * 72;              // private to this wave
#pragma unroll
    for (int ni = 0; ni < 4; ++ni)
#pragma unroll
      for (int r = 0; r < 8; ++r)
        sPw[(roff + r) * 72 + ni * 16 + cis] = (_Float16)sacc[ni][r];
    // same-wave LDS ops are in-order; loads below alias sP so order is kept

    // ---- O += P V ----
#pragma unroll
    for (int ks = 0; ks < 2; ++ks) {
      v16h pf = load_frag(sPw + ks * 32, 72);
#pragma unroll
      for (int no = 0; no < 4; ++no) {
        v16h vf = load_frag(sVt + no * 16 * 72 + ks * 32, 72);
        acc_o[no] = __builtin_amdgcn_wmma_f32_16x16x32_f16(
            false, pf, false, vf, (short)0, acc_o[no], false, false);
      }
    }
  }

  // ---- normalize + write f16 attention output [B*S][1024] ----
#pragma unroll
  for (int r = 0; r < 8; ++r) {
    float inv = 1.0f / l_run[r];
    size_t orow = brow0 + qt * 128 + wave * 16 + roff + r;
#pragma unroll
    for (int no = 0; no < 4; ++no)
      attn[orow * DIM_MODEL + h * DH + no * 16 + cis] = (_Float16)(acc_o[no][r] * inv);
  }
}

// ---------------------------------------------------------------------------
extern "C" void kernel_launch(void* const* d_in, const int* in_sizes, int n_in,
                              void* d_out, int out_size, void* d_ws, size_t ws_size,
                              hipStream_t stream) {
  const float* x      = (const float*)d_in[0];   // [2,2048,1024]
  const float* W_qkv  = (const float*)d_in[1];   // [3072,1024]
  const float* b_qkv  = (const float*)d_in[2];   // [3072]
  const float* W_out  = (const float*)d_in[3];   // [1024,1024]
  const float* b_out  = (const float*)d_in[4];   // [1024]
  float* out = (float*)d_out;

  const int M = 2 * SEQ;                 // 4096 rows
  // workspace layout (f16), 48 MiB total
  char* ws = (char*)d_ws;
  _Float16* xh    = (_Float16*)(ws);                               // 4096*1024
  _Float16* wqkvh = (_Float16*)(ws + (size_t)8  * 1024 * 1024);    // 3072*1024
  _Float16* wouth = (_Float16*)(ws + (size_t)14 * 1024 * 1024);    // 1024*1024
  _Float16* qkvh  = (_Float16*)(ws + (size_t)16 * 1024 * 1024);    // 4096*3072
  _Float16* attnh = (_Float16*)(ws + (size_t)40 * 1024 * 1024);    // 4096*1024

  // 1) fp32 -> f16 conversions
  {
    int n4;
    n4 = (M * DIM_MODEL) / 4;
    cvt_f32_f16<<<(n4 + 255) / 256, 256, 0, stream>>>(x, xh, n4);
    n4 = (3 * DIM_MODEL * DIM_MODEL) / 4;
    cvt_f32_f16<<<(n4 + 255) / 256, 256, 0, stream>>>(W_qkv, wqkvh, n4);
    n4 = (DIM_MODEL * DIM_MODEL) / 4;
    cvt_f32_f16<<<(n4 + 255) / 256, 256, 0, stream>>>(W_out, wouth, n4);
  }

  // 2) QKV projection: [4096,1024] @ [3072,1024]^T + b -> f16 [4096,3072]
  {
    dim3 grid(3 * DIM_MODEL / 128, M / 128);
    gemm_bias<_Float16><<<grid, 256, 0, stream>>>(xh, wqkvh, b_qkv, qkvh,
                                                  M, 3 * DIM_MODEL, DIM_MODEL);
  }

  // 3) causal flash attention -> f16 [4096,1024]
  {
    dim3 grid(SEQ / 128, HEADS, 2);
    attn_flash<<<grid, 256, 0, stream>>>(qkvh, attnh);
  }

  // 4) output projection: [4096,1024] @ [1024,1024]^T + b -> fp32 d_out
  {
    dim3 grid(DIM_MODEL / 128, M / 128);
    gemm_bias<float><<<grid, 256, 0, stream>>>(attnh, wouth, b_out, out,
                                               M, DIM_MODEL, DIM_MODEL);
  }
}
